// Net_63196148793446
// MI455X (gfx1250) — compile-verified, matrix operand order checked by reading, not run on hardware
//
#include <hip/hip_runtime.h>
#include <hip/hip_bf16.h>

// ---------------------------------------------------------------------------
// GIN (5-layer) forward for MI455X (gfx1250).
// GEMMs: v_wmma_f32_16x16x32_f16, wave32; W staged to LDS via async-to-LDS
// (gfx1250 ASYNCcnt path). Scatter/pool: fp32 atomics (node tensor 51MB <
// 192MB L2, so edge traffic is L2-resident).
// ---------------------------------------------------------------------------

typedef __attribute__((ext_vector_type(16))) _Float16 v16h;
typedef __attribute__((ext_vector_type(8)))  float    v8f;
typedef __attribute__((ext_vector_type(4)))  int      v4i;

#define DCH 128   // hidden dim
#define THR 256

#if defined(__gfx1250__) && __has_builtin(__builtin_amdgcn_global_load_async_to_lds_b128)
#define HAVE_ASYNC_LDS 1
#else
#define HAVE_ASYNC_LDS 0
#endif

__device__ __forceinline__ void wait_asynccnt0() {
#if defined(__gfx1250__)
#if __has_builtin(__builtin_amdgcn_s_wait_asynccnt)
    __builtin_amdgcn_s_wait_asynccnt(0);
#else
    asm volatile("s_wait_asynccnt 0x0" ::: "memory");
#endif
#endif
}

// ---------------- elementwise / utility kernels ----------------

__global__ void k_zero_f32(float* __restrict__ p, unsigned n) {
    unsigned i = blockIdx.x * blockDim.x + threadIdx.x;
    if (i < n) p[i] = 0.0f;
}

// scatter-add over edges, 128-feature rows, 4 floats per thread
__global__ void k_scatter128(const float* __restrict__ h, const int* __restrict__ src,
                             const int* __restrict__ dst, float* __restrict__ agg, int E) {
    unsigned idx = blockIdx.x * blockDim.x + threadIdx.x;
    if (idx >= (unsigned)E * 32u) return;
    unsigned e = idx >> 5, c = idx & 31u;
    const float4 v = *(const float4*)(h + (size_t)src[e] * DCH + c * 4);
    float* d = agg + (size_t)dst[e] * DCH + c * 4;
    atomicAdd(d + 0, v.x); atomicAdd(d + 1, v.y);
    atomicAdd(d + 2, v.z); atomicAdd(d + 3, v.w);
}

// generic scatter-add (layer 0: F_IN = 11)
__global__ void k_scatter_gen(const float* __restrict__ h, const int* __restrict__ src,
                              const int* __restrict__ dst, float* __restrict__ agg,
                              int E, int F) {
    unsigned idx = blockIdx.x * blockDim.x + threadIdx.x;
    if (idx >= (unsigned)E * (unsigned)F) return;
    unsigned e = idx / (unsigned)F, f = idx - e * (unsigned)F;
    atomicAdd(agg + (size_t)dst[e] * F + f, h[(size_t)src[e] * F + f]);
}

// out_h[row, k] = f16( h[row,k] + agg[row,k] ) zero-padded to Kpad = 1<<kshift
__global__ void k_addcvt_pad(const float* __restrict__ h, const float* __restrict__ agg,
                             _Float16* __restrict__ out, int M, int Fin, int kshift,
                             unsigned total) {
    unsigned idx = blockIdx.x * blockDim.x + threadIdx.x;
    if (idx >= total) return;
    unsigned row = idx >> kshift;
    unsigned k   = idx & ((1u << kshift) - 1u);
    float v = 0.0f;
    if ((int)row < M && (int)k < Fin) {
        size_t o = (size_t)row * Fin + k;
        v = h[o] + agg[o];
    }
    out[idx] = (_Float16)v;
}

// Wt[n*Kpad + k] = f16(W[k*128 + n]) zero-padded in k  (transposed so B-frag
// 16-byte groups are contiguous -> ds_load_b128)
__global__ void k_wcvt_pad(const float* __restrict__ W, _Float16* __restrict__ Wt,
                           int Kin, int kshift) {
    unsigned idx = blockIdx.x * blockDim.x + threadIdx.x;
    unsigned total = 128u << kshift;
    if (idx >= total) return;
    unsigned n = idx >> kshift, k = idx & ((1u << kshift) - 1u);
    Wt[idx] = (k < (unsigned)Kin) ? (_Float16)W[(size_t)k * DCH + n] : (_Float16)0.0f;
}

// ---------------- WMMA GEMM: out = relu(A @ W + bias) ----------------
// A:    Mtiles*16 x KPAD f16 row-major; Wt: 128 x KPAD f16 (transposed W)
// OF -> write fp32 M x 128; OH -> write f16 M x 128 (compile-time selection)
// block = 128 threads = 4 waves; wave computes 16 rows x 128 cols.
template <int KPAD, bool OF, bool OH>
__global__ void k_gemm_wmma(const _Float16* __restrict__ A, const _Float16* __restrict__ Wt,
                            const float* __restrict__ bias,
                            float* __restrict__ outF, _Float16* __restrict__ outH,
                            int Mtiles) {
    __shared__ uint32_t lw32[(DCH / 2) * KPAD];        // 128 x KPAD f16
    constexpr int wBytes = DCH * KPAD * 2;

    // ---- stage W into LDS ----
#if HAVE_ASYNC_LDS
    {
        const char* gsrc = (const char*)Wt;
        char*       ldst = (char*)lw32;
        for (int ofs = threadIdx.x * 16; ofs < wBytes; ofs += 128 * 16) {
            __builtin_amdgcn_global_load_async_to_lds_b128(
                (v4i __attribute__((address_space(1)))*)(gsrc + ofs),
                (v4i __attribute__((address_space(3)))*)(ldst + ofs), 0, 0);
        }
        wait_asynccnt0();
    }
#else
    {
        const uint32_t* gw = (const uint32_t*)Wt;
        for (int i = threadIdx.x; i < wBytes / 4; i += 128) lw32[i] = gw[i];
    }
#endif
    __syncthreads();

    const int wave = threadIdx.x >> 5;
    const int lane = threadIdx.x & 31;
    int tile = blockIdx.x * 4 + wave;
    if (tile >= Mtiles) tile = Mtiles - 1;             // wave-uniform clamp (dup work ok)
    const int m    = lane & 15;
    const int half = lane >> 4;

    const _Float16* arow = A + (size_t)(tile * 16 + m) * KPAD;
    const _Float16* lw   = (const _Float16*)lw32;
    __builtin_prefetch(arow, 0, 3);                    // global_prefetch_b8

    union Frag { uint32_t u[8]; uint4 q[2]; v16h h; };

    v8f acc[8];
    #pragma unroll
    for (int t = 0; t < 8; ++t) acc[t] = (v8f){0.f,0.f,0.f,0.f,0.f,0.f,0.f,0.f};

    constexpr int KSTEPS = KPAD >> 5;
    Frag af[2];
    {   // first A fragment
        const int kb = half * 8;
        af[0].q[0] = *(const uint4*)(arow + kb);
        af[0].q[1] = *(const uint4*)(arow + kb + 16);
    }
    #pragma unroll
    for (int ks = 0; ks < KSTEPS; ++ks) {
        if (ks + 1 < KSTEPS) {                         // double-buffer next A frag
            const int kb = (ks + 1) * 32 + half * 8;
            af[(ks + 1) & 1].q[0] = *(const uint4*)(arow + kb);
            af[(ks + 1) & 1].q[1] = *(const uint4*)(arow + kb + 16);
        }
        // preload ALL 8 B fragments (2x ds_load_b128 each) before the WMMA burst
        Frag bf[8];
        const int kb = ks * 32 + half * 8;
        #pragma unroll
        for (int nt = 0; nt < 8; ++nt) {
            const _Float16* bcol = lw + (size_t)(nt * 16 + m) * KPAD;
            bf[nt].q[0] = *(const uint4*)(bcol + kb);
            bf[nt].q[1] = *(const uint4*)(bcol + kb + 16);
        }
        #pragma unroll
        for (int nt = 0; nt < 8; ++nt)
            acc[nt] = __builtin_amdgcn_wmma_f32_16x16x32_f16(
                false, af[ks & 1].h, false, bf[nt].h, (short)0, acc[nt], false, false);
    }

    // epilogue: bias + relu; C layout: VGPR r -> row = r + 8*half, col = lane&15
    #pragma unroll
    for (int nt = 0; nt < 8; ++nt) {
        const int col = nt * 16 + m;
        const float bv = bias[col];
        #pragma unroll
        for (int r = 0; r < 8; ++r) {
            float v = acc[nt][r] + bv;
            v = v > 0.0f ? v : 0.0f;
            const size_t o = (size_t)(tile * 16 + r + half * 8) * DCH + col;
            if constexpr (OF) outF[o] = v;
            if constexpr (OH) outH[o] = (_Float16)v;
        }
    }
}

// ---------------- batchnorm (coalesced two-stage) ----------------

// stage 1: row-major coalesced partial sums; thread t owns column t&127
__global__ void k_bn_partial(const float* __restrict__ z, float* __restrict__ sums, int Nn) {
    __shared__ float s1[THR], s2[THR];
    const int col = threadIdx.x & (DCH - 1);
    const int rp  = threadIdx.x >> 7;                  // 0 or 1
    float a = 0.f, b = 0.f;
    for (int i = blockIdx.x * 2 + rp; i < Nn; i += gridDim.x * 2) {
        float v = z[(size_t)i * DCH + col];
        a += v; b += v * v;
    }
    s1[threadIdx.x] = a; s2[threadIdx.x] = b;
    __syncthreads();
    if (rp == 0) {
        atomicAdd(&sums[col],       s1[threadIdx.x] + s1[threadIdx.x + 128]);
        atomicAdd(&sums[DCH + col], s2[threadIdx.x] + s2[threadIdx.x + 128]);
    }
}

__global__ void k_bn_finalize(const float* __restrict__ sums, float* __restrict__ stats, int Nn) {
    int f = threadIdx.x;                               // 128 threads
    float mu  = sums[f] / Nn;
    float var = sums[DCH + f] / Nn - mu * mu;
    stats[f]       = mu;
    stats[DCH + f] = rsqrtf(var + 1e-5f);
}

__global__ void k_bn_apply(const float* __restrict__ z, const float* __restrict__ stats,
                           const float* __restrict__ gamma, const float* __restrict__ beta,
                           float* __restrict__ h, unsigned n) {
    unsigned i = blockIdx.x * blockDim.x + threadIdx.x;
    if (i >= n) return;
    unsigned f = i & (DCH - 1);
    h[i] = (z[i] - stats[f]) * stats[DCH + f] * gamma[f] + beta[f];
}

// ---------------- pooling / fc glue ----------------

__global__ void k_pool(const float* __restrict__ h, const int* __restrict__ bb,
                       float* __restrict__ pooled, int Nn) {
    unsigned idx = blockIdx.x * blockDim.x + threadIdx.x;
    if (idx >= (unsigned)Nn * 32u) return;
    unsigned node = idx >> 5, c = idx & 31u;
    int g = bb[node];
    const float4 v = *(const float4*)(h + (size_t)node * DCH + c * 4);
    float* d = pooled + (size_t)g * DCH + c * 4;
    atomicAdd(d + 0, v.x); atomicAdd(d + 1, v.y);
    atomicAdd(d + 2, v.z); atomicAdd(d + 3, v.w);
}

__global__ void k_fc_prep(const float* __restrict__ acc, const float* __restrict__ pooled,
                          _Float16* __restrict__ sh, int G, unsigned total) {
    unsigned i = blockIdx.x * blockDim.x + threadIdx.x;
    if (i >= total) return;
    unsigned row = i >> 7;
    float v = 0.0f;
    if ((int)row < G) v = acc[i] + pooled[i];
    sh[i] = (_Float16)v;
}

__global__ void k_acc_add(float* __restrict__ acc, const float* __restrict__ g, unsigned n) {
    unsigned i = blockIdx.x * blockDim.x + threadIdx.x;
    if (i < n) acc[i] += g[i];
}

__global__ void k_final_dot(const float* __restrict__ g, const float* __restrict__ W,
                            const float* __restrict__ b, float* __restrict__ out, int Gn) {
    int i = blockIdx.x * blockDim.x + threadIdx.x;
    if (i >= Gn) return;
    const float* row = g + (size_t)i * DCH;
    float s = 0.f;
    #pragma unroll 8
    for (int k = 0; k < DCH; ++k) s += row[k] * W[k];
    out[i] = s + b[0];
}

// ---------------------------------------------------------------------------

extern "C" void kernel_launch(void* const* d_in, const int* in_sizes, int n_in,
                              void* d_out, int out_size, void* d_ws, size_t ws_size,
                              hipStream_t stream) {
    const int F_IN = 11;
    const float* x   = (const float*)d_in[0];
    const int*   ei  = (const int*)d_in[1];
    const int*   bat = (const int*)d_in[2];
    const int N = in_sizes[0] / F_IN;
    const int E = in_sizes[1] / 2;
    const int G = out_size;
    const int* src = ei;
    const int* dst = ei + E;

    // params flattening: detect jax sorted-dict vs insertion order per conv
    // sorted:    beta, gamma, l1W, l1b, l2W, l2b   (in_sizes[3] == 128)
    // insertion: l1W, l1b, l2W, l2b, gamma, beta   (in_sizes[3] == 11*128)
    const bool sortedOrder = (in_sizes[3] == DCH);
    auto convP = [&](int c, int which) -> const float* {
        // which: 0=beta 1=gamma 2=l1W 3=l1b 4=l2W 5=l2b
        int base = 3 + c * 6;
        if (sortedOrder) return (const float*)d_in[base + which];
        const int map[6] = {5, 4, 0, 1, 2, 3};
        return (const float*)d_in[base + map[which]];
    };
    const int fcBase = 3 + 5 * 6;                      // 33
    auto fcW = [&](int f) { return (const float*)d_in[fcBase + f * 2]; };
    auto fcB = [&](int f) { return (const float*)d_in[fcBase + f * 2 + 1]; };
    const float* lin2W = (const float*)d_in[fcBase + 10];
    const float* lin2B = (const float*)d_in[fcBase + 11];

    const int Ntiles = (N + 15) / 16, Np = Ntiles * 16;
    const int Gtiles = (G + 15) / 16, Gp = Gtiles * 16;

    // workspace carve-up
    char* base = (char*)d_ws;
    size_t off = 0;
    auto alloc = [&](size_t bytes) { void* p = base + off; off = (off + bytes + 255) & ~(size_t)255; return p; };
    float*    agg    = (float*)   alloc((size_t)Np * DCH * 4);  // reused as z2
    float*    hbuf   = (float*)   alloc((size_t)Np * DCH * 4);
    _Float16* aggh   = (_Float16*)alloc((size_t)Np * DCH * 2);
    _Float16* z1h    = (_Float16*)alloc((size_t)Np * DCH * 2);
    _Float16* wt     = (_Float16*)alloc((size_t)DCH * DCH * 2);
    float*    sums   = (float*)   alloc(2 * DCH * 4);
    float*    stats  = (float*)   alloc(2 * DCH * 4);
    float*    pooled = (float*)   alloc((size_t)G * DCH * 4);
    float*    accb   = (float*)   alloc((size_t)G * DCH * 4);
    _Float16* sh     = (_Float16*)alloc((size_t)Gp * DCH * 2);
    float*    gbuf   = (float*)   alloc((size_t)Gp * DCH * 4);
    (void)ws_size; (void)n_in;
    float* z2 = agg;

    auto nb = [](size_t n) { return dim3((unsigned)((n + THR - 1) / THR)); };

    k_zero_f32<<<nb((size_t)G * DCH), THR, 0, stream>>>(accb, (unsigned)(G * DCH));

    for (int layer = 0; layer < 5; ++layer) {
        const float* hin = (layer == 0) ? x : hbuf;
        const int Fin    = (layer == 0) ? F_IN : DCH;
        const int kshift = (layer == 0) ? 5 : 7;

        // agg = scatter_add(h[src] -> dst)
        k_zero_f32<<<nb((size_t)Np * DCH), THR, 0, stream>>>(agg, (unsigned)(Np * DCH));
        if (layer == 0)
            k_scatter_gen<<<nb((size_t)E * Fin), THR, 0, stream>>>(hin, src, dst, agg, E, Fin);
        else
            k_scatter128<<<nb((size_t)E * 32), THR, 0, stream>>>(hin, src, dst, agg, E);

        // aggh = f16(h + agg), zero-padded in K
        k_addcvt_pad<<<nb((size_t)Np << kshift), THR, 0, stream>>>(
            hin, agg, aggh, N, Fin, kshift, (unsigned)(Np << kshift));

        // z1 = relu(aggh @ W1 + b1) -> f16
        k_wcvt_pad<<<nb((size_t)DCH << kshift), THR, 0, stream>>>(convP(layer, 2), wt, Fin, kshift);
        if (layer == 0)
            k_gemm_wmma<32, false, true><<<dim3((Ntiles + 3) / 4), 128, 0, stream>>>(
                aggh, wt, convP(layer, 3), nullptr, z1h, Ntiles);
        else
            k_gemm_wmma<128, false, true><<<dim3((Ntiles + 3) / 4), 128, 0, stream>>>(
                aggh, wt, convP(layer, 3), nullptr, z1h, Ntiles);

        // z2 = relu(z1 @ W2 + b2) -> fp32 (aliases agg)
        k_wcvt_pad<<<nb((size_t)DCH << 7), THR, 0, stream>>>(convP(layer, 4), wt, DCH, 7);
        k_gemm_wmma<128, true, false><<<dim3((Ntiles + 3) / 4), 128, 0, stream>>>(
            z1h, wt, convP(layer, 5), z2, nullptr, Ntiles);

        // batchnorm -> h
        k_zero_f32<<<dim3(1), THR, 0, stream>>>(sums, 2 * DCH);
        k_bn_partial<<<dim3(256), THR, 0, stream>>>(z2, sums, N);
        k_bn_finalize<<<dim3(1), DCH, 0, stream>>>(sums, stats, N);
        k_bn_apply<<<nb((size_t)N * DCH), THR, 0, stream>>>(
            z2, stats, convP(layer, 1), convP(layer, 0), hbuf, (unsigned)(N * DCH));

        // pooled = segment_sum(h, batch)
        k_zero_f32<<<nb((size_t)G * DCH), THR, 0, stream>>>(pooled, (unsigned)(G * DCH));
        k_pool<<<nb((size_t)N * 32), THR, 0, stream>>>(hbuf, bat, pooled, N);

        // g = relu((acc + pooled) @ Wfc + bfc); acc += g
        k_fc_prep<<<nb((size_t)Gp * DCH), THR, 0, stream>>>(accb, pooled, sh, G,
                                                            (unsigned)(Gp * DCH));
        k_wcvt_pad<<<nb((size_t)DCH << 7), THR, 0, stream>>>(fcW(layer), wt, DCH, 7);
        k_gemm_wmma<128, true, false><<<dim3((Gtiles + 3) / 4), 128, 0, stream>>>(
            sh, wt, fcB(layer), gbuf, nullptr, Gtiles);
        k_acc_add<<<nb((size_t)G * DCH), THR, 0, stream>>>(accb, gbuf, (unsigned)(G * DCH));
    }

    k_final_dot<<<nb((size_t)G), THR, 0, stream>>>(gbuf, lin2W, lin2B, (float*)d_out, G);
}
